// MusicLSTM_2439541424367
// MI455X (gfx1250) — matrix operand order repaired and checked by reading, not compiled, not run on hardware
//
#include <hip/hip_runtime.h>

// MusicLSTM collapsed kernel for MI455X (gfx1250), wave32 + WMMA + async LDS.
// B=131072 batch, E=13 experts, I=13 inputs, H=64 hidden, G=4H=256 gates.
// f-gate (cols 64..127) is dead because c0 = 0.

typedef _Float16 v16h __attribute__((ext_vector_type(16)));
typedef float    v8f  __attribute__((ext_vector_type(8)));

#define BB 131072
#define EE 13
#define II 13
#define HH 64
#define GG 256
#define ROWS_PER_BLOCK 128
#define XS_FLOATS (ROWS_PER_BLOCK * EE * II)          // 21632 floats = 86528 B
#define XS_CHUNKS (XS_FLOATS * 4 / 16)                // 5408 x 16B async chunks
#define ZROW (EE * GG)                                // index of zero pad row in W16

__device__ __forceinline__ float fast_tanh(float x) {
#if __has_builtin(__builtin_amdgcn_tanhf)
  return __builtin_amdgcn_tanhf(x);       // v_tanh_f32 (gfx1250 TRANS op)
#else
  float e2 = __expf(-2.0f * x);
  return (1.0f - e2) / (1.0f + e2);
#endif
}

__device__ __forceinline__ float fast_sigmoid(float x) {
  return 0.5f + 0.5f * fast_tanh(0.5f * x);
}

// Async 16B global -> LDS copy (CDNA5, tracked by ASYNCcnt).
__device__ __forceinline__ void async_copy16(uint32_t lds_off, const void* gptr) {
  asm volatile("global_load_async_to_lds_b128 %0, %1, off"
               :: "v"(lds_off), "v"((unsigned long long)(uintptr_t)gptr)
               : "memory");
}

// ------------------------------------------------------------------
// Pre-pack: W_ih[e, g, 0:13] (f32) -> W16[e*256+g][0:16] (f16, K padded
// to 16 with zeros, 32B rows => one aligned 32-byte B-operand row), plus
// one all-zero pad row (index ZROW) that the hi half-wave loads so B's
// K=16..31 lanes are zero without EXEC divergence.
// biassum = b_ih + b_hh. Tiny: ~120 KB total, stays hot in L2/WGP$.
// ------------------------------------------------------------------
__global__ void pack_weights(const float* __restrict__ W_ih,
                             const float* __restrict__ b_ih,
                             const float* __restrict__ b_hh,
                             _Float16* __restrict__ W16,
                             float* __restrict__ biassum) {
  const int stride = gridDim.x * blockDim.x;
  const int tid = blockIdx.x * blockDim.x + threadIdx.x;
  for (int i = tid; i < (ZROW + 1) * 16; i += stride) {
    const int k = i & 15;
    const int row = i >> 4;                 // e*256 + gate_col  (ZROW = pad)
    const bool valid = (row < ZROW) && (k < II);
    W16[i] = valid ? (_Float16)W_ih[row * II + k] : (_Float16)0.0f;
  }
  for (int i = tid; i < EE * GG; i += stride)
    biassum[i] = b_ih[i] + b_hh[i];
}

// ------------------------------------------------------------------
// Main kernel: 256 threads = 8 waves per block; each wave owns 16 batch
// rows. Block stages its contiguous x slice into LDS with async b128
// copies. Per expert: build A tile (16x32 f16) from LDS, then for each
// hidden chunk of 16: WMMA (C = inline 0) for i/g/o gates, bias add +
// activations + W_lin scaling in the epilogue, cross-lane reduction over
// the 16 hidden columns, sigmoid, store.
// ------------------------------------------------------------------
__global__ __launch_bounds__(256) void lstm_wmma(
    const float* __restrict__ x,            // [B, E, I]
    const _Float16* __restrict__ W16,       // [E*256+1, 16] f16 padded
    const float* __restrict__ biassum,      // [E*256]
    const float* __restrict__ W_lin,        // [E, H]
    const float* __restrict__ b_lin,        // [E]
    float* __restrict__ out) {              // [B, E]
  extern __shared__ float xs[];             // ROWS_PER_BLOCK*169 floats (+pad)

  const int lane = threadIdx.x & 31;
  const int wave = threadIdx.x >> 5;
  const int row_base = blockIdx.x * ROWS_PER_BLOCK + wave * 16;
  const int n = lane & 15;                  // N column (B/C/D); row M for A
  const bool hi = lane >= 16;

  // ---- stage x slice for this block: fully coalesced async 16B chunks.
  {
    const uint32_t lds0 = (uint32_t)(uintptr_t)xs;
    const char* gbase =
        (const char*)(x + (size_t)blockIdx.x * (ROWS_PER_BLOCK * EE * II));
    for (int i = threadIdx.x; i < XS_CHUNKS; i += blockDim.x)
      async_copy16(lds0 + 16u * (uint32_t)i, gbase + 16 * (size_t)i);
    asm volatile("s_wait_asynccnt 0x0" ::: "memory");
  }
  __syncthreads();

  const int lrow = wave * 16 + n;           // local row within block
  const int kbase = hi ? 8 : 0;

  // Branchless per-lane B addressing (byte offsets vs uniform W16 base):
  // lo lanes: off = (e*GG + col) * 32 ; hi lanes: off = ZROW * 32 (zero row).
  const uint32_t sel  = hi ? 0u : ~0u;                 // mask for lo lanes
  const uint32_t hib  = ~sel & ((uint32_t)ZROW << 5);  // pad-row byte offset
  const uint32_t nb   = sel & ((uint32_t)n << 5);      // per-lane column part
  const char* wbase   = (const char*)W16;

#pragma unroll 1
  for (int e = 0; e < EE; ++e) {
    // ---- A operand (16x32 f16, contiguous K order matching B):
    // lanes 0-15 : elems 0..15 = K0..15
    // lanes 16-31: elems 0..15 = K16..31 (zero)
    // K=13..15 read junk from LDS; harmless: B rows K>=13 are zero.
    const float* xrow = &xs[lrow * (EE * II) + e * II];
    v16h amat = {};
#pragma unroll
    for (int j = 0; j < 16; ++j)
      amat[j] = hi ? (_Float16)0.0f : (_Float16)xrow[j];

    const uint32_t ebase = ((uint32_t)(e * GG) << 5) & sel;  // expert byte base

    float acc[8];
#pragma unroll
    for (int r = 0; r < 8; ++r) acc[r] = 0.0f;

#pragma unroll
    for (int c4 = 0; c4 < 4; ++c4) {        // hidden chunk: cols 16*c4..+15
      v8f d[3];
      const int gate_base[3] = {0, 128, 192};   // i, g, o  (f is dead)
#pragma unroll
      for (int gi = 0; gi < 3; ++gi) {
        // B operand: lanes 0-15 load the padded K0..15 row (32B aligned),
        // lanes 16-31 load the shared zero pad row (K16..31 == 0).
        const uint32_t cb = ((uint32_t)((gate_base[gi] + c4 * 16) << 5)) & sel;
        const uint32_t off = (ebase + cb + nb) | hib;
        const v16h bmat = *(const v16h*)(wbase + off);
        v8f zc = {};                        // folds to inline SRC2=0
        d[gi] = __builtin_amdgcn_wmma_f32_16x16x32_f16(
            false, amat, false, bmat, (short)0, zc, false, false);
      }

      // ---- bias + activations + W_lin scaling; lane holds hidden idx
      // 16*c4+n for rows r (lanes 0-15) / r+8 (lanes 16-31).
      const int bcol = c4 * 16 + n;
      const float bi = biassum[e * GG + 0   + bcol];
      const float bg = biassum[e * GG + 128 + bcol];
      const float bo = biassum[e * GG + 192 + bcol];
      const float wl = W_lin[e * HH + bcol];
#pragma unroll
      for (int r = 0; r < 8; ++r) {
        const float ig = d[0][r] + bi;
        const float gg = d[1][r] + bg;
        const float og = d[2][r] + bo;
        const float cv = fast_sigmoid(ig) * fast_tanh(gg);
        const float hv = fast_sigmoid(og) * fast_tanh(cv);
        acc[r] += hv * wl;
      }
    }

    // ---- reduce the 16 hidden columns across each half-wave.
#pragma unroll
    for (int r = 0; r < 8; ++r) {
      float v = acc[r];
      v += __shfl_xor(v, 1, 32);
      v += __shfl_xor(v, 2, 32);
      v += __shfl_xor(v, 4, 32);
      v += __shfl_xor(v, 8, 32);
      acc[r] = v;
    }
    const float bl = b_lin[e];
    if (n == 0) {                           // lane 0 -> rows 0..7, lane 16 -> 8..15
      const int rowoff = row_base + (hi ? 8 : 0);
#pragma unroll
      for (int r = 0; r < 8; ++r)
        out[(size_t)(rowoff + r) * EE + e] = fast_sigmoid(acc[r] + bl);
    }
  }
}

extern "C" void kernel_launch(void* const* d_in, const int* in_sizes, int n_in,
                              void* d_out, int out_size, void* d_ws, size_t ws_size,
                              hipStream_t stream) {
  const float* x     = (const float*)d_in[0];
  const float* W_ih  = (const float*)d_in[1];
  // d_in[2] = W_hh — unused (h0 == 0).
  const float* b_ih  = (const float*)d_in[3];
  const float* b_hh  = (const float*)d_in[4];
  const float* W_lin = (const float*)d_in[5];
  const float* b_lin = (const float*)d_in[6];
  float* out = (float*)d_out;

  // Workspace: (13*256+1)*16 f16 (106528 B) + 13*256 f32 (13312 B) ≈ 120 KB.
  _Float16* W16  = (_Float16*)d_ws;
  float* biassum = (float*)((char*)d_ws + (size_t)(ZROW + 1) * 16 * sizeof(_Float16));

  pack_weights<<<64, 256, 0, stream>>>(W_ih, b_ih, b_hh, W16, biassum);

  const size_t lds_bytes = (size_t)XS_FLOATS * sizeof(float) + 16;  // +OOB pad
  lstm_wmma<<<BB / ROWS_PER_BLOCK, 256, lds_bytes, stream>>>(
      x, W16, biassum, W_lin, b_lin, out);
}
